// GRUSVS_gs_31095563223399
// MI455X (gfx1250) — matrix operand order repaired and checked by reading, not compiled, not run on hardware
//
#include <hip/hip_runtime.h>
#include <stdint.h>

// ---------------------------------------------------------------------------
// GRU seq2seq (encoder BiGRU + attention decoder) for MI455X / gfx1250.
// All GEMMs via v_wmma_f32_16x16x32_bf16 (wave32). Weights pre-packed into
// the WMMA B-operand lane layout; activations bf16 row-major, A packed on the
// fly. Sequential recurrences run as persistent 32-WG kernels with a
// device-scope generation barrier. Working set is L2-resident (192MB L2).
// New this round: encoder input projection gi = x@wih + bih hoisted out of
// the serialized recurrence into a whole-GPU parallel GEMM (ws permitting).
// ---------------------------------------------------------------------------

typedef __bf16 bf16x16 __attribute__((ext_vector_type(16)));
typedef float  v8f     __attribute__((ext_vector_type(8)));

#define DEV __device__ __forceinline__
#define Z8 {0.f,0.f,0.f,0.f,0.f,0.f,0.f,0.f}

constexpr int Bz = 32, Sz = 512, Tz = 512, Dz = 512;
constexpr int KXz = 544;            // 514 padded to 17*32
constexpr int Gz  = 1536;           // 3*D
constexpr int OUTz = 1324;

constexpr int KT_WIH = 17, KT_WHH = 16, NT_G = 96;
constexpr int KT_FCE = 32, NT_FCE = 32;    // fcenc  K=1024 N=512
constexpr int KT_ATT = 32, NT_ATT = 32;    // attn(enc) K=1024 N=512
constexpr int KT_DWIH = 64, KT_DWHH = 16;  // dec GRU K=2048/512 N=1536
constexpr int KT_FCH = 80, NT_FCH = 64;    // fchid K=2560 N=1024
constexpr int KT_FCO = 32, NT_FCO = 83;    // fcout K=1024 N=1328(pad)

DEV uint16_t f2bf(float f) {
  uint32_t u = __float_as_uint(f);
  u += 0x7fffu + ((u >> 16) & 1u);
  return (uint16_t)(u >> 16);
}
DEV float bf2f(uint32_t h) { return __uint_as_float(h << 16); }
DEV float sigm(float x) { return 1.f / (1.f + __expf(-x)); }
DEV float tanh_f(float x) {
  float e = __expf(-2.f * fabsf(x));
  float t = (1.f - e) / (1.f + e);
  return copysignf(t, x);
}

union ABu { uint32_t u[8]; bf16x16 v; };

// A operand (16x32 bf16 tile): per ISA 7.12.2, lane m=lane&15,
// v0..3 -> K = 8*hi + {0,2,4,6}, v4..7 -> K = 16 + 8*hi + {0,2,4,6}.
DEV void load_a(const uint16_t* base, int lda, ABu& a) {
  int lane = threadIdx.x & 31;
  const uint16_t* r = base + (size_t)(lane & 15) * lda + ((lane >> 4) << 3);
  a.u[0] = *(const uint32_t*)(r + 0);
  a.u[1] = *(const uint32_t*)(r + 2);
  a.u[2] = *(const uint32_t*)(r + 4);
  a.u[3] = *(const uint32_t*)(r + 6);
  a.u[4] = *(const uint32_t*)(r + 16);
  a.u[5] = *(const uint32_t*)(r + 18);
  a.u[6] = *(const uint32_t*)(r + 20);
  a.u[7] = *(const uint32_t*)(r + 22);
}
// B operand: pre-packed, lane's 8 dwords contiguous (2x global_load_b128).
DEV void load_b(const uint32_t* tile, ABu& b) {
  const uint32_t* p = tile + ((threadIdx.x & 31) << 3);
#pragma unroll
  for (int v = 0; v < 8; ++v) b.u[v] = p[v];
}
DEV v8f wmma_bf(const ABu& a, const ABu& b, v8f c) {
  return __builtin_amdgcn_wmma_f32_16x16x32_bf16(false, a.v, false, b.v,
                                                 (short)0, c, false, false);
}

// Device-scope generation barrier (persistent-grid sync).
DEV void gbar(unsigned* bar, unsigned nwg) {
  __threadfence();
  __syncthreads();
  if (threadIdx.x == 0) {
    unsigned g = __hip_atomic_load(&bar[1], __ATOMIC_RELAXED, __HIP_MEMORY_SCOPE_AGENT);
    unsigned prev = __hip_atomic_fetch_add(&bar[0], 1u, __ATOMIC_ACQ_REL, __HIP_MEMORY_SCOPE_AGENT);
    if (prev == nwg - 1u) {
      __hip_atomic_store(&bar[0], 0u, __ATOMIC_RELAXED, __HIP_MEMORY_SCOPE_AGENT);
      __hip_atomic_fetch_add(&bar[1], 1u, __ATOMIC_RELEASE, __HIP_MEMORY_SCOPE_AGENT);
    } else {
      while (__hip_atomic_load(&bar[1], __ATOMIC_ACQUIRE, __HIP_MEMORY_SCOPE_AGENT) == g)
        __builtin_amdgcn_s_sleep(2);
    }
  }
  __syncthreads();
}

// --------------------------- prep: sort + init -----------------------------
__global__ __launch_bounds__(256) void k_prep(const int* length, int* idx, int* lens,
                                              unsigned* bars, float* h_f, float* h_b,
                                              uint16_t* hbf0, uint16_t* hbb0) {
  __shared__ int L[Bz];
  int tid = threadIdx.x;
  if (tid < Bz) L[tid] = length[tid];
  __syncthreads();
  if (tid < Bz) {
    int li = L[tid], rank = 0;
    for (int jj = 0; jj < Bz; ++jj) {
      int lj = L[jj];
      if (lj > li || (lj == li && jj < tid)) rank++;
    }
    idx[rank] = tid;
    lens[rank] = li;
  }
  if (tid < 4) bars[tid] = 0u;
  for (int i = tid; i < Bz * Dz; i += 256) {
    h_f[i] = 0.f; h_b[i] = 0.f; hbf0[i] = 0; hbb0[i] = 0;
  }
}

// -------------------- pack fp32 weights -> WMMA B tiles --------------------
// dst[((nt*KT+kt)*32+lane)*8+v] = pack(W[k0,n], W[k0+1,n]) bf16 pair,
// n = nt*16 + (lane&15), k0 = kt*32 + (lane>=16?8:0) + (v>=4?16:0) + 2(v&3). remap: fchid K-perm.
__global__ __launch_bounds__(256) void k_pack(const float* W, int Ksrc, int Nsrc,
                                              int KT, int NT, uint32_t* dst, int remap) {
  size_t total = (size_t)KT * NT * 256;
  for (size_t i = (size_t)blockIdx.x * blockDim.x + threadIdx.x; i < total;
       i += (size_t)gridDim.x * blockDim.x) {
    int v = (int)(i & 7);
    int lane = (int)((i >> 3) & 31);
    size_t tile = i >> 8;
    int kt = (int)(tile % KT), nt = (int)(tile / KT);
    int n = nt * 16 + (lane & 15);
    int k0 = kt * 32 + ((lane & 16) ? 8 : 0) + ((v & 4) ? 16 : 0) + 2 * (v & 3);
    uint32_t lo = 0, hi = 0;
    if (n < Nsrc) {
      int ka = k0, kb = k0 + 1;
      if (remap) {  // ours [h,inp,w] -> src [h,w,inp]
        ka = (ka < 512) ? ka : (ka < 1536 ? ka + 1024 : ka - 1024);
        kb = (kb < 512) ? kb : (kb < 1536 ? kb + 1024 : kb - 1024);
      }
      if (k0     < Ksrc) lo = f2bf(W[(size_t)ka * Nsrc + n]);
      if (k0 + 1 < Ksrc) hi = f2bf(W[(size_t)kb * Nsrc + n]);
    }
    dst[i] = lo | (hi << 16);
  }
}

// hid_w (attn_w[:512]) transposed to bf16: dst[n*512+k] = W[k*512+n]
__global__ __launch_bounds__(256) void k_trans(const float* W, uint16_t* dst) {
  for (int i = blockIdx.x * 256 + threadIdx.x; i < Dz * Dz; i += gridDim.x * 256) {
    int n = i >> 9, k = i & 511;
    dst[i] = f2bf(W[k * Dz + n]);
  }
}

// -------- x = concat(emb[phone[idx]], beats[idx], pitch[idx]) as bf16 ------
__global__ __launch_bounds__(256) void k_xbuild(const float* emb, const int* phone,
                                                const float* pitch, const float* beats,
                                                const int* idx, uint16_t* xb) {
  size_t total = (size_t)Sz * Bz * KXz;
  for (size_t i = (size_t)blockIdx.x * blockDim.x + threadIdx.x; i < total;
       i += (size_t)gridDim.x * blockDim.x) {
    int k = (int)(i % KXz);
    size_t r = i / KXz;
    int b = (int)(r % Bz), s = (int)(r / Bz);
    int sb = idx[b];
    float val = 0.f;
    if (k < 512) {
      int ph = phone[sb * Sz + s];
      val = emb[(size_t)ph * 512 + k];
    } else if (k == 512) val = beats[sb * Sz + s];
    else if (k == 513)   val = pitch[sb * Sz + s];
    xb[i] = f2bf(val);
  }
}

// ------------------------ persistent BiGRU encoder -------------------------
// 32 WGs x 192 thr: WG 0..15 fwd, 16..31 bwd; WG j owns 32 hidden columns.
// 6 waves: {r0,r1,z0,z1,n0,n1} gate tiles; n-waves keep i_n/h_n separate.
// hb (bf16 GEMM input) ping-pongs -> one device barrier per step.
// use_gi=1: gi = x@wih + bih precomputed in parallel; only h@whh is serialized.
__global__ __launch_bounds__(192) void k_enc(const uint16_t* xb,
    const uint32_t* pk_wih_f, const uint32_t* pk_wih_b,
    const uint32_t* pk_whh_f, const uint32_t* pk_whh_b,
    const float* bih_f, const float* bhh_f, const float* bih_b, const float* bhh_b,
    const float* gi_f, const float* gi_b, int use_gi,
    const int* lens, float* h_f, float* h_b,
    uint16_t* hbf0, uint16_t* hbf1, uint16_t* hbb0, uint16_t* hbb1,
    uint16_t* enc_b, unsigned* bar) {
  __shared__ float Lr[32][32], Lz[32][32], Lin[32][32], Lhn[32][32];
  __shared__ float eb[6][32];
  __shared__ int slen[32];
  int wg = blockIdx.x, bwd = wg >> 4, j = wg & 15;
  const uint32_t* wih = bwd ? pk_wih_b : pk_wih_f;
  const uint32_t* whh = bwd ? pk_whh_b : pk_whh_f;
  const float* bih = bwd ? bih_b : bih_f;
  const float* bhh = bwd ? bhh_b : bhh_f;
  const float* gi  = bwd ? gi_b : gi_f;
  float* h = bwd ? h_b : h_f;
  uint16_t* hp0 = bwd ? hbb0 : hbf0;
  uint16_t* hp1 = bwd ? hbb1 : hbf1;
  int dirofs = bwd ? Dz : 0;
  int tid = threadIdx.x, wave = tid >> 5, lane = tid & 31;
  int pair = wave & 1, gate = wave >> 1;  // 0=r 1=z 2=n
  int ntl = gate * 32 + 2 * j + pair;
  const uint32_t* wihT = wih + (size_t)ntl * KT_WIH * 256;
  const uint32_t* whhT = whh + (size_t)ntl * KT_WHH * 256;

  if (tid < 32) {
    slen[tid] = lens[tid];
    int col = 32 * j + tid;
    eb[0][tid] = bih[col] + bhh[col];                 // r full (no-gi path)
    eb[1][tid] = bih[Dz + col] + bhh[Dz + col];       // z full (no-gi path)
    eb[2][tid] = bih[2 * Dz + col];                   // i_n (no-gi path)
    eb[3][tid] = bhh[2 * Dz + col];                   // h_n
    eb[4][tid] = bhh[col];                            // r hh-only (gi path)
    eb[5][tid] = bhh[Dz + col];                       // z hh-only (gi path)
  }
  __syncthreads();

  for (int step = 0; step < Sz; ++step) {
    int s = bwd ? (Sz - 1 - step) : step;
    const uint16_t* Ax = xb + (size_t)s * Bz * KXz;
    const uint16_t* cur = (step & 1) ? hp1 : hp0;
    uint16_t* nxt = (step & 1) ? hp0 : hp1;
    v8f c0 = Z8, c1 = Z8, d0 = Z8, d1 = Z8;
    if (!use_gi) {
      // x @ wih (K padded to 544; pad cols are zero on both sides)
      for (int kt = 0; kt < KT_WIH; ++kt) {
        ABu bb, a0, a1;
        load_b(wihT + (size_t)kt * 256, bb);
        load_a(Ax + kt * 32, KXz, a0);
        load_a(Ax + 16 * KXz + kt * 32, KXz, a1);
        c0 = wmma_bf(a0, bb, c0);
        c1 = wmma_bf(a1, bb, c1);
      }
    }
    // h @ whh (n-gate keeps this separate for r*(h_n+bhh))
    for (int kt = 0; kt < KT_WHH; ++kt) {
      ABu bb, a0, a1;
      load_b(whhT + (size_t)kt * 256, bb);
      load_a(cur + kt * 32, Dz, a0);
      load_a(cur + 16 * Dz + kt * 32, Dz, a1);
      if (gate == 2) { d0 = wmma_bf(a0, bb, d0); d1 = wmma_bf(a1, bb, d1); }
      else           { c0 = wmma_bf(a0, bb, c0); c1 = wmma_bf(a1, bb, c1); }
    }
    {
      int cc = pair * 16 + (lane & 15);
      int mb = (lane >> 4) << 3;
      float (*dst)[32] = (gate == 0) ? Lr : (gate == 1) ? Lz : Lin;
#pragma unroll
      for (int v = 0; v < 8; ++v) { dst[mb + v][cc] = c0[v]; dst[16 + mb + v][cc] = c1[v]; }
      if (gate == 2) {
#pragma unroll
        for (int v = 0; v < 8; ++v) { Lhn[mb + v][cc] = d0[v]; Lhn[16 + mb + v][cc] = d1[v]; }
      }
    }
    __syncthreads();
    for (int e = tid; e < 1024; e += 192) {
      int m = e >> 5, cc = e & 31, col = 32 * j + cc;
      float gr, gzv, gin;
      float ghn = Lhn[m][cc] + eb[3][cc];
      if (use_gi) {
        const float* gr_row = gi + ((size_t)s * Bz + m) * Gz;
        gr  = Lr[m][cc] + gr_row[col] + eb[4][cc];
        gzv = Lz[m][cc] + gr_row[Dz + col] + eb[5][cc];
        gin = gr_row[2 * Dz + col];                 // bih already folded in
      } else {
        gr  = Lr[m][cc] + eb[0][cc];
        gzv = Lz[m][cc] + eb[1][cc];
        gin = Lin[m][cc] + eb[2][cc];
      }
      float rr = sigm(gr), zz = sigm(gzv);
      float nn = tanh_f(gin + rr * ghn);
      float hold = h[m * Dz + col];
      float hcand = (1.f - zz) * nn + zz * hold;
      bool valid = s < slen[m];
      float hnew = valid ? hcand : hold;
      h[m * Dz + col] = hnew;
      nxt[m * Dz + col] = f2bf(hnew);
      enc_b[((size_t)m * Sz + s) * (2 * Dz) + dirofs + col] = valid ? f2bf(hcand) : (uint16_t)0;
    }
    gbar(bar, 32);
  }
}

// hcat = bf16 concat(hf, hb)
__global__ __launch_bounds__(256) void k_hcat(const float* h_f, const float* h_b, uint16_t* hcat) {
  for (int i = blockIdx.x * 256 + threadIdx.x; i < Bz * 2 * Dz; i += gridDim.x * 256) {
    int b = i >> 10, k = i & 1023;
    float v = (k < Dz) ? h_f[b * Dz + k] : h_b[b * Dz + k - Dz];
    hcat[i] = f2bf(v);
  }
}

// hidden = tanh(hcat @ fcenc + b); init decoder state (hid f32, zall hid slot,
// zero the inp slot).
__global__ __launch_bounds__(64) void k_fcenc(const uint16_t* hcat, const uint32_t* pk,
                                              const float* bias, float* hid, uint16_t* zall) {
  int nt = blockIdx.x, wave = threadIdx.x >> 5, lane = threadIdx.x & 31;
  const uint16_t* A = hcat + (size_t)wave * 16 * (2 * Dz);
  const uint32_t* BT = pk + (size_t)nt * KT_FCE * 256;
  v8f c = Z8;
  for (int kt = 0; kt < KT_FCE; ++kt) {
    ABu a, bb;
    load_a(A + kt * 32, 2 * Dz, a);
    load_b(BT + (size_t)kt * 256, bb);
    c = wmma_bf(a, bb, c);
  }
  int col = nt * 16 + (lane & 15);
  int mb = wave * 16 + ((lane >> 4) << 3);
#pragma unroll
  for (int v = 0; v < 8; ++v) {
    float t = tanh_f(c[v] + bias[col]);
    hid[(mb + v) * Dz + col] = t;
    zall[(mb + v) * 2560 + col] = f2bf(t);
  }
  for (int e = threadIdx.x; e < 1024; e += 64) {
    int m = e >> 5, cc = e & 31;
    zall[m * 2560 + Dz + nt * 32 + cc] = 0;  // inp0 = 0
  }
}

// --------------------- generic parallel GEMM (bf16->f32) -------------------
__global__ __launch_bounds__(256) void k_gemm(const uint16_t* A, int lda,
                                              const uint32_t* Bpk, int KT, int NT,
                                              float* C, int ldc, int Nbound,
                                              const float* bias, int Mtiles) {
  int wid = blockIdx.x * 8 + (threadIdx.x >> 5);
  int lane = threadIdx.x & 31;
  int nw = gridDim.x * 8;
  int tiles = Mtiles * NT;
  for (int tdx = wid; tdx < tiles; tdx += nw) {
    int mt = tdx / NT, nt = tdx % NT;
    const uint16_t* At = A + (size_t)mt * 16 * lda;
    const uint32_t* BT = Bpk + (size_t)nt * KT * 256;
    v8f c = Z8;
    for (int kt = 0; kt < KT; ++kt) {
      if (kt + 1 < KT) __builtin_prefetch(BT + (size_t)(kt + 1) * 256, 0, 1);
      ABu a, bb;
      load_a(At + kt * 32, lda, a);
      load_b(BT + (size_t)kt * 256, bb);
      c = wmma_bf(a, bb, c);
    }
    int n = nt * 16 + (lane & 15);
    if (n < Nbound) {
      float bv = bias[n];
      int mb = mt * 16 + ((lane >> 4) << 3);
#pragma unroll
      for (int v = 0; v < 8; ++v) C[(size_t)(mb + v) * ldc + n] = c[v] + bv;
    }
  }
}

// ----------------------- persistent attention decoder ----------------------
// 32 WGs x 256 thr. Per step: S1 attention (WG b handles batch b),
// S2 GRU gates (WG j owns 16 hidden cols), S3 fchid (WG j owns 2 N-tiles).
// zall row = bf16 [hid(512) | inp(1024) | weighted(1024)], stride 2560.
__global__ __launch_bounds__(256) void k_dec(const float* encpart, const uint16_t* enc_b,
    const uint16_t* hwT, const float* v_w,
    const uint32_t* pk_dwih, const uint32_t* pk_dwhh,
    const float* dbih, const float* dbhh,
    const uint32_t* pk_fchid, const float* fchid_b,
    const int* lens, float* hid, uint16_t* zall, uint16_t* outbuf, unsigned* bar) {
  __shared__ float qv[512], hrow[512], vv[512], sc[512], red[256];
  __shared__ float Lp[8][32][16];
  __shared__ float cb[4][16], fb[32];
  __shared__ int slen[32];
  int b = blockIdx.x, j = b;
  int tid = threadIdx.x, wave = tid >> 5, lane = tid & 31;

  // one-time staging of per-step constants
  for (int i = tid; i < 512; i += 256) vv[i] = v_w[i];
  if (tid < 32) { slen[tid] = lens[tid]; fb[tid] = fchid_b[32 * j + tid]; }
  if (tid < 16) {
    int col = 16 * j + tid;
    cb[0][tid] = dbih[col] + dbhh[col];
    cb[1][tid] = dbih[Dz + col] + dbhh[Dz + col];
    cb[2][tid] = dbih[2 * Dz + col];
    cb[3][tid] = dbhh[2 * Dz + col];
  }
  __syncthreads();

  for (int t = 0; t < Tz; ++t) {
    // ---- S1: attention for batch b ----
    for (int i = tid; i < 512; i += 256) hrow[i] = hid[b * Dz + i];
    __syncthreads();
    for (int c = tid; c < 512; c += 256) {  // q = hid[b] @ hid_w
      const uint16_t* wr = hwT + (size_t)c * Dz;
      float acc = 0.f;
      for (int k = 0; k < Dz; k += 8) {
        uint4 w = *(const uint4*)(wr + k);
        acc += hrow[k + 0] * bf2f(w.x & 0xffff) + hrow[k + 1] * bf2f(w.x >> 16)
             + hrow[k + 2] * bf2f(w.y & 0xffff) + hrow[k + 3] * bf2f(w.y >> 16)
             + hrow[k + 4] * bf2f(w.z & 0xffff) + hrow[k + 5] * bf2f(w.z >> 16)
             + hrow[k + 6] * bf2f(w.w & 0xffff) + hrow[k + 7] * bf2f(w.w >> 16);
      }
      qv[c] = acc;
    }
    __syncthreads();
    for (int si = wave; si < Sz; si += 8) {  // scores
      const float* ep = encpart + ((size_t)b * Sz + si) * Dz;
      float p = 0.f;
      for (int d = lane; d < Dz; d += 32) p += tanh_f(qv[d] + ep[d]) * vv[d];
#pragma unroll
      for (int m = 16; m; m >>= 1) p += __shfl_xor(p, m, 32);
      if (lane == 0) sc[si] = p;
    }
    __syncthreads();
    float mx = -3.4e38f;                      // softmax over S
    for (int i = tid; i < Sz; i += 256) mx = fmaxf(mx, sc[i]);
    red[tid] = mx; __syncthreads();
    for (int off = 128; off; off >>= 1) { if (tid < off) red[tid] = fmaxf(red[tid], red[tid + off]); __syncthreads(); }
    mx = red[0]; __syncthreads();
    float sm = 0.f;
    for (int i = tid; i < Sz; i += 256) { float e = __expf(sc[i] - mx); sc[i] = e; sm += e; }
    red[tid] = sm; __syncthreads();
    for (int off = 128; off; off >>= 1) { if (tid < off) red[tid] += red[tid + off]; __syncthreads(); }
    float inv = 1.f / red[0]; __syncthreads();
    for (int i = tid; i < Sz; i += 256) sc[i] *= inv;
    __syncthreads();
    {  // weighted = aw @ enc_out[b] : 4 cols per thread
      int d0 = tid * 4;
      float a0 = 0, a1 = 0, a2 = 0, a3 = 0;
      const uint16_t* eb2 = enc_b + (size_t)b * Sz * (2 * Dz) + d0;
      for (int s2 = 0; s2 < Sz; ++s2) {
        float awv = sc[s2];
        uint2 pv = *(const uint2*)(eb2 + (size_t)s2 * (2 * Dz));
        a0 += awv * bf2f(pv.x & 0xffff); a1 += awv * bf2f(pv.x >> 16);
        a2 += awv * bf2f(pv.y & 0xffff); a3 += awv * bf2f(pv.y >> 16);
      }
      uint16_t* zw = zall + (size_t)b * 2560 + 1536 + d0;
      zw[0] = f2bf(a0); zw[1] = f2bf(a1); zw[2] = f2bf(a2); zw[3] = f2bf(a3);
    }
    gbar(bar, 32);  // weighted visible everywhere

    // ---- S2: GRU gates (reads zall hid_old + [inp|weighted]) ----
    {
      int gate = wave & 3, half = wave >> 2;  // 0=r 1=z 2=i_n 3=h_n
      v8f c0 = Z8, c1 = Z8;
      if (gate <= 1) {
        int nt = (gate == 0) ? j : (32 + j);
        const uint32_t* BW = pk_dwih + (size_t)nt * KT_DWIH * 256;
        const uint32_t* BH = pk_dwhh + (size_t)nt * KT_DWHH * 256;
        for (int kk = half * 40; kk < half * 40 + 40; ++kk) {
          ABu a0, a1, bb;
          if (kk < 64) {
            load_b(BW + (size_t)kk * 256, bb);
            const uint16_t* A = zall + Dz;
            load_a(A + kk * 32, 2560, a0);
            load_a(A + 16 * 2560 + kk * 32, 2560, a1);
          } else {
            int kt = kk - 64;
            load_b(BH + (size_t)kt * 256, bb);
            load_a(zall + kt * 32, 2560, a0);
            load_a(zall + 16 * 2560 + kt * 32, 2560, a1);
          }
          c0 = wmma_bf(a0, bb, c0);
          c1 = wmma_bf(a1, bb, c1);
        }
      } else if (gate == 2) {
        int nt = 64 + j;
        const uint32_t* BW = pk_dwih + (size_t)nt * KT_DWIH * 256;
        const uint16_t* A = zall + Dz;
        for (int kt = half * 32; kt < half * 32 + 32; ++kt) {
          ABu a0, a1, bb;
          load_b(BW + (size_t)kt * 256, bb);
          load_a(A + kt * 32, 2560, a0);
          load_a(A + 16 * 2560 + kt * 32, 2560, a1);
          c0 = wmma_bf(a0, bb, c0);
          c1 = wmma_bf(a1, bb, c1);
        }
      } else {
        int nt = 64 + j;
        const uint32_t* BH = pk_dwhh + (size_t)nt * KT_DWHH * 256;
        for (int kt = half * 8; kt < half * 8 + 8; ++kt) {
          ABu a0, a1, bb;
          load_b(BH + (size_t)kt * 256, bb);
          load_a(zall + kt * 32, 2560, a0);
          load_a(zall + 16 * 2560 + kt * 32, 2560, a1);
          c0 = wmma_bf(a0, bb, c0);
          c1 = wmma_bf(a1, bb, c1);
        }
      }
      int cc = lane & 15, mb = (lane >> 4) << 3;
#pragma unroll
      for (int v = 0; v < 8; ++v) { Lp[wave][mb + v][cc] = c0[v]; Lp[wave][16 + mb + v][cc] = c1[v]; }
    }
    gbar(bar, 32);  // all reads of zall hid done -> safe to overwrite
    for (int e = tid; e < 512; e += 256) {
      int m = e >> 4, cc = e & 15, col = 16 * j + cc;
      float gr  = Lp[0][m][cc] + Lp[4][m][cc] + cb[0][cc];
      float gzv = Lp[1][m][cc] + Lp[5][m][cc] + cb[1][cc];
      float gin = Lp[2][m][cc] + Lp[6][m][cc] + cb[2][cc];
      float ghn = Lp[3][m][cc] + Lp[7][m][cc] + cb[3][cc];
      float rr = sigm(gr), zz = sigm(gzv);
      float nn = tanh_f(gin + rr * ghn);
      float hold = hid[m * Dz + col];
      float hnew = (1.f - zz) * nn + zz * hold;
      hid[m * Dz + col] = hnew;
      zall[m * 2560 + col] = f2bf(hnew);
    }
    gbar(bar, 32);  // new hid visible for fchid

    // ---- S3: pred = relu([hid_n, weighted, inp] @ fchid + b) ----
    {
      int ntl = wave >> 2, q = wave & 3;
      int nt = 2 * j + ntl;
      const uint32_t* BT = pk_fchid + (size_t)nt * KT_FCH * 256;
      v8f c0 = Z8, c1 = Z8;
      for (int kt = q * 20; kt < q * 20 + 20; ++kt) {
        ABu a0, a1, bb;
        load_b(BT + (size_t)kt * 256, bb);
        load_a(zall + kt * 32, 2560, a0);
        load_a(zall + 16 * 2560 + kt * 32, 2560, a1);
        c0 = wmma_bf(a0, bb, c0);
        c1 = wmma_bf(a1, bb, c1);
      }
      int cc = lane & 15, mb = (lane >> 4) << 3;
#pragma unroll
      for (int v = 0; v < 8; ++v) { Lp[wave][mb + v][cc] = c0[v]; Lp[wave][16 + mb + v][cc] = c1[v]; }
    }
    gbar(bar, 32);  // all reads of zall inp done -> safe to overwrite
    for (int e = tid; e < 1024; e += 256) {
      int th = e >> 9, m = (e >> 4) & 31, cc = e & 15;
      int n = (2 * j + th) * 16 + cc;
      float val = Lp[th * 4][m][cc] + Lp[th * 4 + 1][m][cc] +
                  Lp[th * 4 + 2][m][cc] + Lp[th * 4 + 3][m][cc] + fb[th * 16 + cc];
      float p = fmaxf(val, 0.f);
      zall[m * 2560 + Dz + n] = f2bf(p);  // inp for next step (unmasked)
      outbuf[((size_t)m * Tz + t) * (2 * Dz) + n] = (t < slen[m]) ? f2bf(p) : (uint16_t)0;
    }
    // next-step consumers of pred are gated by the first barrier of step t+1
  }
}

// ---------------------------------------------------------------------------
extern "C" void kernel_launch(void* const* d_in, const int* in_sizes, int n_in,
                              void* d_out, int out_size, void* d_ws, size_t ws_size,
                              hipStream_t stream) {
  (void)in_sizes; (void)n_in; (void)out_size;
  const int*   phone   = (const int*)d_in[1];
  const float* pitch   = (const float*)d_in[2];
  const float* beats   = (const float*)d_in[3];
  const int*   length  = (const int*)d_in[4];
  const float* emb     = (const float*)d_in[6];
  const float* ewih_f  = (const float*)d_in[7];
  const float* ewhh_f  = (const float*)d_in[8];
  const float* ebih_f  = (const float*)d_in[9];
  const float* ebhh_f  = (const float*)d_in[10];
  const float* ewih_b  = (const float*)d_in[11];
  const float* ewhh_b  = (const float*)d_in[12];
  const float* ebih_b  = (const float*)d_in[13];
  const float* ebhh_b  = (const float*)d_in[14];
  const float* fcenc_w = (const float*)d_in[15];
  const float* fcenc_b = (const float*)d_in[16];
  const float* attn_w  = (const float*)d_in[17];
  const float* attn_b  = (const float*)d_in[18];
  const float* v_w     = (const float*)d_in[19];
  const float* dwih    = (const float*)d_in[20];
  const float* dwhh    = (const float*)d_in[21];
  const float* dbih    = (const float*)d_in[22];
  const float* dbhh    = (const float*)d_in[23];
  const float* fchid_w = (const float*)d_in[24];
  const float* fchid_b = (const float*)d_in[25];
  const float* fcout_w = (const float*)d_in[26];
  const float* fcout_b = (const float*)d_in[27];

  char* ws = (char*)d_ws;
  size_t off = 0;
  auto alloc = [&](size_t bytes) { size_t r = off; off += (bytes + 255) & ~(size_t)255; return r; };

  unsigned* bars    = (unsigned*)(ws + alloc(16));
  int* idx          = (int*)(ws + alloc(Bz * 4));
  int* lens         = (int*)(ws + alloc(Bz * 4));
  uint16_t* xb      = (uint16_t*)(ws + alloc((size_t)Sz * Bz * KXz * 2));
  uint32_t* pwih_f  = (uint32_t*)(ws + alloc((size_t)KT_WIH * NT_G * 1024));
  uint32_t* pwih_b  = (uint32_t*)(ws + alloc((size_t)KT_WIH * NT_G * 1024));
  uint32_t* pwhh_f  = (uint32_t*)(ws + alloc((size_t)KT_WHH * NT_G * 1024));
  uint32_t* pwhh_b  = (uint32_t*)(ws + alloc((size_t)KT_WHH * NT_G * 1024));
  uint32_t* pfce    = (uint32_t*)(ws + alloc((size_t)KT_FCE * NT_FCE * 1024));
  uint32_t* patt    = (uint32_t*)(ws + alloc((size_t)KT_ATT * NT_ATT * 1024));
  uint16_t* hwT     = (uint16_t*)(ws + alloc((size_t)Dz * Dz * 2));
  uint32_t* pdwih   = (uint32_t*)(ws + alloc((size_t)KT_DWIH * NT_G * 1024));
  uint32_t* pdwhh   = (uint32_t*)(ws + alloc((size_t)KT_DWHH * NT_G * 1024));
  uint32_t* pfchid  = (uint32_t*)(ws + alloc((size_t)KT_FCH * NT_FCH * 1024));
  uint32_t* pfcout  = (uint32_t*)(ws + alloc((size_t)KT_FCO * NT_FCO * 1024));
  float* h_f        = (float*)(ws + alloc((size_t)Bz * Dz * 4));
  float* h_b        = (float*)(ws + alloc((size_t)Bz * Dz * 4));
  uint16_t* hbf0    = (uint16_t*)(ws + alloc((size_t)Bz * Dz * 2));
  uint16_t* hbf1    = (uint16_t*)(ws + alloc((size_t)Bz * Dz * 2));
  uint16_t* hbb0    = (uint16_t*)(ws + alloc((size_t)Bz * Dz * 2));
  uint16_t* hbb1    = (uint16_t*)(ws + alloc((size_t)Bz * Dz * 2));
  uint16_t* enc_b   = (uint16_t*)(ws + alloc((size_t)Bz * Sz * 2 * Dz * 2));
  float* encpart    = (float*)(ws + alloc((size_t)Bz * Sz * Dz * 4));
  uint16_t* hcat    = (uint16_t*)(ws + alloc((size_t)Bz * 2 * Dz * 2));
  float* hid        = (float*)(ws + alloc((size_t)Bz * Dz * 4));
  uint16_t* zall    = (uint16_t*)(ws + alloc((size_t)Bz * 2560 * 2));
  uint16_t* outbuf  = (uint16_t*)(ws + alloc((size_t)Bz * Tz * 2 * Dz * 2));

  // Optional: hoist gi = x@wih + bih out of the serialized encoder loop.
  // Needs 2 x 100.7MB fp32; only enabled if the workspace is big enough.
  size_t gi_bytes = (size_t)Sz * Bz * Gz * 4;
  float* gi_f = nullptr;
  float* gi_b = nullptr;
  int use_gi = 0;
  if (ws_size >= off + 2 * (gi_bytes + 256)) {
    gi_f = (float*)(ws + alloc(gi_bytes));
    gi_b = (float*)(ws + alloc(gi_bytes));
    use_gi = 1;
  } else {
    gi_f = (float*)(ws);  // unused (guarded by use_gi), keep non-null
    gi_b = (float*)(ws);
  }

  // phase 0: sort, barrier/state init
  k_prep<<<1, 256, 0, stream>>>(length, idx, lens, bars, h_f, h_b, hbf0, hbb0);

  // phase 1: weight packing (one-time per launch), hid_w transpose, x build
  k_pack<<<256, 256, 0, stream>>>(ewih_f, 514, Gz, KT_WIH, NT_G, pwih_f, 0);
  k_pack<<<256, 256, 0, stream>>>(ewih_b, 514, Gz, KT_WIH, NT_G, pwih_b, 0);
  k_pack<<<256, 256, 0, stream>>>(ewhh_f, 512, Gz, KT_WHH, NT_G, pwhh_f, 0);
  k_pack<<<256, 256, 0, stream>>>(ewhh_b, 512, Gz, KT_WHH, NT_G, pwhh_b, 0);
  k_pack<<<256, 256, 0, stream>>>(fcenc_w, 1024, Dz, KT_FCE, NT_FCE, pfce, 0);
  k_pack<<<256, 256, 0, stream>>>(attn_w + (size_t)Dz * Dz, 1024, Dz, KT_ATT, NT_ATT, patt, 0);
  k_pack<<<256, 256, 0, stream>>>(dwih, 2048, Gz, KT_DWIH, NT_G, pdwih, 0);
  k_pack<<<256, 256, 0, stream>>>(dwhh, 512, Gz, KT_DWHH, NT_G, pdwhh, 0);
  k_pack<<<256, 256, 0, stream>>>(fchid_w, 2560, 1024, KT_FCH, NT_FCH, pfchid, 1);
  k_pack<<<256, 256, 0, stream>>>(fcout_w, 1024, OUTz, KT_FCO, NT_FCO, pfcout, 0);
  k_trans<<<512, 256, 0, stream>>>(attn_w, hwT);
  k_xbuild<<<4096, 256, 0, stream>>>(emb, phone, pitch, beats, idx, xb);

  // phase 1b: parallel input projection for both encoder directions
  if (use_gi) {
    k_gemm<<<512, 256, 0, stream>>>(xb, KXz, pwih_f, KT_WIH, NT_G,
                                    gi_f, Gz, Gz, ebih_f, (Sz * Bz) / 16);
    k_gemm<<<512, 256, 0, stream>>>(xb, KXz, pwih_b, KT_WIH, NT_G,
                                    gi_b, Gz, Gz, ebih_b, (Sz * Bz) / 16);
  }

  // phase 2: persistent BiGRU encoder (1 device barrier / step)
  k_enc<<<32, 192, 0, stream>>>(xb, pwih_f, pwih_b, pwhh_f, pwhh_b,
                                ebih_f, ebhh_f, ebih_b, ebhh_b,
                                gi_f, gi_b, use_gi,
                                lens, h_f, h_b, hbf0, hbf1, hbb0, hbb1,
                                enc_b, bars);

  // phase 3: decoder init + enc_part (L2-resident parallel GEMM)
  k_hcat<<<32, 256, 0, stream>>>(h_f, h_b, hcat);
  k_fcenc<<<32, 64, 0, stream>>>(hcat, pfce, fcenc_b, hid, zall);
  k_gemm<<<512, 256, 0, stream>>>(enc_b, 2 * Dz, patt, KT_ATT, NT_ATT,
                                  encpart, Dz, Dz, attn_b, (Bz * Sz) / 16);

  // phase 4: persistent attention decoder (4 device barriers / step)
  k_dec<<<32, 256, 0, stream>>>(encpart, enc_b, hwT, v_w, pdwih, pdwhh,
                                dbih, dbhh, pfchid, fchid_b,
                                lens, hid, zall, outbuf, bars + 2);

  // phase 5: final projection outs @ fcout + b -> d_out (fp32, B*T x 1324)
  k_gemm<<<512, 256, 0, stream>>>(outbuf, 2 * Dz, pfcout, KT_FCO, NT_FCO,
                                  (float*)d_out, OUTz, OUTz, fcout_b, (Bz * Tz) / 16);
}
